// CNNFeatMoe_83562883711000
// MI455X (gfx1250) — matrix operand order repaired and checked by reading, not compiled, not run on hardware
//
#include <hip/hip_runtime.h>
#include <hip/hip_bf16.h>
#include <stdint.h>

typedef __bf16 bf16;
typedef __attribute__((ext_vector_type(16))) __bf16 v16bf;
typedef __attribute__((ext_vector_type(8)))  __bf16 v8bf;
typedef __attribute__((ext_vector_type(8)))  float  v8f;

// ---------------- workspace layout (bytes, all 256-aligned) ----------------
// a1    : bf16 [4][256][32][32][32]  conv1 out (relu+pool), NHWC (c fastest)
// h2    : bf16 [4][256][16384]       conv2 out (relu+pool), NCHW flat = FC K order
// bmat2 : bf16 [4][9][64][32]        conv2 weights, (e,kb,n,c) K-fastest per column
// gpool : bf16 [256][65536]          gate conv out (relu+pool), NCHW flat
// feat  : f32  [4][256][128]         expert FC outputs (atomic-accumulated)
// logits: f32  [256][4]
static const size_t OFF_A1    = 0;
static const size_t OFF_H2    = 67108864ull;   // 64 MiB after a1
static const size_t OFF_BMAT2 = 100663296ull;  // +32 MiB
static const size_t OFF_GPOOL = 100810752ull;  // +144 KiB
static const size_t OFF_FEAT  = 134365184ull;  // +32 MiB
static const size_t OFF_LOG   = 134889472ull;  // +512 KiB

// ---------------- expert conv1: 3->32, relu, 2x2 maxpool (direct, fp32) ----
__global__ __launch_bounds__(256)
void moe_expert_conv1(const float* __restrict__ x, const float* __restrict__ ew1,
                      const float* __restrict__ eb1, bf16* __restrict__ a1) {
  int l = blockIdx.x * 256 + threadIdx.x;      // 4*256*32*32*32 threads
  const int c   = l & 31; l >>= 5;
  const int pxp = l & 31; l >>= 5;
  const int pyp = l & 31; l >>= 5;
  const int b   = l & 255; l >>= 8;
  const int e   = l;
  float wl[27];
#pragma unroll
  for (int i = 0; i < 27; ++i) wl[i] = ew1[(size_t)(e * 32 + c) * 27 + i];
  const float bias = eb1[e * 32 + c];
  float pooled = 0.f;
#pragma unroll
  for (int iy = 0; iy < 2; ++iy)
#pragma unroll
    for (int ix = 0; ix < 2; ++ix) {
      const int y = pyp * 2 + iy, xq = pxp * 2 + ix;
      float s = bias;
#pragma unroll
      for (int ci = 0; ci < 3; ++ci)
#pragma unroll
        for (int ky = 0; ky < 3; ++ky) {
          const int yy = y + ky - 1;
          if (yy < 0 || yy >= 64) continue;
#pragma unroll
          for (int kx = 0; kx < 3; ++kx) {
            const int xx = xq + kx - 1;
            if (xx < 0 || xx >= 64) continue;
            s += x[((size_t)(b * 3 + ci) * 64 + yy) * 64 + xx] * wl[ci * 9 + ky * 3 + kx];
          }
        }
      pooled = fmaxf(pooled, fmaxf(s, 0.f));
    }
  a1[(((size_t)(e * 256 + b) * 32 + pyp) * 32 + pxp) * 32 + c] = (bf16)pooled;
}

// ---------------- gate conv: 3->64, relu, 2x2 maxpool (direct, fp32) -------
__global__ __launch_bounds__(256)
void moe_gate_conv(const float* __restrict__ x, const float* __restrict__ gw,
                   const float* __restrict__ gb, bf16* __restrict__ gpool) {
  int l = blockIdx.x * 256 + threadIdx.x;      // 256*64*32*32 threads
  const int pxp = l & 31; l >>= 5;
  const int pyp = l & 31; l >>= 5;
  const int c   = l & 63; l >>= 6;
  const int b   = l;
  float wl[27];
#pragma unroll
  for (int i = 0; i < 27; ++i) wl[i] = gw[(size_t)c * 27 + i];
  const float bias = gb[c];
  float pooled = 0.f;
#pragma unroll
  for (int iy = 0; iy < 2; ++iy)
#pragma unroll
    for (int ix = 0; ix < 2; ++ix) {
      const int y = pyp * 2 + iy, xq = pxp * 2 + ix;
      float s = bias;
#pragma unroll
      for (int ci = 0; ci < 3; ++ci)
#pragma unroll
        for (int ky = 0; ky < 3; ++ky) {
          const int yy = y + ky - 1;
          if (yy < 0 || yy >= 64) continue;
#pragma unroll
          for (int kx = 0; kx < 3; ++kx) {
            const int xx = xq + kx - 1;
            if (xx < 0 || xx >= 64) continue;
            s += x[((size_t)(b * 3 + ci) * 64 + yy) * 64 + xx] * wl[ci * 9 + ky * 3 + kx];
          }
        }
      pooled = fmaxf(pooled, fmaxf(s, 0.f));
    }
  gpool[(size_t)b * 65536 + c * 1024 + pyp * 32 + pxp] = (bf16)pooled;
}

// --------- prep conv2 weights into (e, kb, n, c) bf16 B-matrix layout ------
__global__ __launch_bounds__(256)
void moe_prep_b2(const float* __restrict__ ew2, bf16* __restrict__ bmat2) {
  int idx = blockIdx.x * 256 + threadIdx.x;    // 4*9*64*32 = 73728 threads
  const int c = idx & 31; idx >>= 5;
  const int n = idx & 63; idx >>= 6;
  const int kb = idx % 9;
  const int e  = idx / 9;
  const int ky = kb / 3, kx = kb - ky * 3;
  bmat2[((size_t)(e * 9 + kb) * 64 + n) * 32 + c] =
      (bf16)ew2[(((size_t)(e * 64 + n) * 32 + c) * 3 + ky) * 3 + kx];
}

// -------- conv2 implicit GEMM (M=262144, N=64, K=288) via bf16 WMMA --------
// M ordering: m = (((b*16+py)*16+px)*2+iy)*2+ix  -> 4 consecutive rows form
// one 2x2 pool window; relu+pool happens in per-lane accumulator registers.
// Whole 288x64 B operand (36 KB) is staged into LDS ONCE; the k-loop then
// runs barrier-free so global A loads pipeline across all 9 unrolled steps.
__global__ __launch_bounds__(256)
void moe_conv2_wmma(const bf16* __restrict__ a1, const bf16* __restrict__ bmat2,
                    const float* __restrict__ eb2, bf16* __restrict__ h2) {
  __shared__ __align__(64) bf16 Bs[9 * 64 * 32];  // [kb][n][kk], 36 KB
  const int t    = threadIdx.x;
  const int lane = t & 31;
  const int wave = t >> 5;
  const int e    = blockIdx.y;
  const int m_base = (blockIdx.x * 8 + wave) * 16;
  const int row  = m_base + (lane & 15);
  const int ixo  = row & 1, iyo = (row >> 1) & 1;
  const int px   = (row >> 2) & 15, py = (row >> 6) & 15;
  const int bb   = row >> 10;
  const int y    = py * 2 + iyo, xq = px * 2 + ixo;
  const int kh   = (lane >> 4) * 16;           // A-frag K-half per lane group
  const int bcol = lane & 15;
  const int bks  = (lane >> 4) * 16;           // B-frag K-half per lane group

  // stage all 9 B tiles once: 36864 B = 2304 uint4, 9 per thread, coalesced
  {
    const uint4* src = (const uint4*)(bmat2 + (size_t)e * 18432);
#pragma unroll
    for (int i = 0; i < 9; ++i) ((uint4*)Bs)[t + i * 256] = src[t + i * 256];
  }
  __syncthreads();

  v8f acc[4];
#pragma unroll
  for (int nt = 0; nt < 4; ++nt) acc[nt] = v8f{};

#pragma unroll
  for (int kb = 0; kb < 9; ++kb) {
    const int ky = kb / 3, kx = kb - ky * 3;
    const int yy = y + ky - 1, xx = xq + kx - 1;
    v8bf lo = v8bf{}, hi = v8bf{};
    if (yy >= 0 && yy < 32 && xx >= 0 && xx < 32) {
      const bf16* ap = a1 + (((size_t)((e * 256 + bb) * 32 + yy) * 32 + xx) * 32);
      lo = *(const v8bf*)(ap + kh);
      hi = *(const v8bf*)(ap + kh + 8);
    }
    v16bf af;
#pragma unroll
    for (int i = 0; i < 8; ++i) { af[i] = lo[i]; af[8 + i] = hi[i]; }

#pragma unroll
    for (int nt = 0; nt < 4; ++nt) {
      v16bf bfrag = *(const v16bf*)&Bs[((kb * 64) + nt * 16 + bcol) * 32 + bks];
      acc[nt] = __builtin_amdgcn_wmma_f32_16x16x32_bf16(
          false, af, false, bfrag, (short)0, acc[nt], false, false);
    }
  }

  // epilogue: bias + relu + 2x2 maxpool entirely within each lane's 8 regs
  const int wbase = (m_base >> 2) + (lane >> 4) * 2;  // pool-window index
#pragma unroll
  for (int nt = 0; nt < 4; ++nt) {
    const int n = nt * 16 + (lane & 15);
    const float bias = eb2[e * 64 + n];
    float v[8];
#pragma unroll
    for (int i = 0; i < 8; ++i) v[i] = fmaxf(acc[nt][i] + bias, 0.f);
    const float p0 = fmaxf(fmaxf(v[0], v[1]), fmaxf(v[2], v[3]));
    const float p1 = fmaxf(fmaxf(v[4], v[5]), fmaxf(v[6], v[7]));
    const int w0 = wbase, w1 = wbase + 1;       // w = b*256 + py*16 + px
    h2[(size_t)(e * 256 + (w0 >> 8)) * 16384 + n * 256 + (w0 & 255)] = (bf16)p0;
    h2[(size_t)(e * 256 + (w1 >> 8)) * 16384 + n * 256 + (w1 & 255)] = (bf16)p1;
  }
}

// ---- expert FC GEMM (M=256, N=128, K=16384) bf16 WMMA, 16-way K-split -----
// The WG's whole A chunk (16 rows x 1024 K = 32 KB) is staged into LDS once;
// the 32-step k-loop is barrier-free: {global B load + prefetch, cvt, wmma}.
__global__ __launch_bounds__(256)
void moe_fc_wmma(const bf16* __restrict__ h2, const float* __restrict__ efw,
                 float* __restrict__ feat) {
  __shared__ __align__(64) bf16 As[16 * 1024];   // [row][k], 32 KB
  const int t = threadIdx.x, lane = t & 31, wave = t >> 5;
  const int kc = blockIdx.x;                   // K-chunk (16 x 1024)
  const int m0 = blockIdx.y * 16;
  const int e  = blockIdx.z;
  const int n0 = wave * 16;

  // stage A chunk: 2048 uint4, 8 per thread; consecutive threads are
  // consecutive 16B words within a row -> fully coalesced
  {
#pragma unroll
    for (int i = 0; i < 8; ++i) {
      const int idx = t + i * 256;             // uint4 index, row = idx/128
      const int r   = idx >> 7;
      const int col = idx & 127;               // 16B word within row
      ((uint4*)As)[idx] = *(const uint4*)(
          h2 + (size_t)(e * 256 + m0 + r) * 16384 + kc * 1024 + col * 8);
    }
  }
  __syncthreads();

  const int arow = lane & 15, akh = (lane >> 4) * 16;
  const int bn   = n0 + (lane & 15);
  const float* brow =
      efw + (size_t)(e * 128 + bn) * 16384 + kc * 1024 + (lane >> 4) * 16;

  v8f acc = v8f{};
  for (int it = 0; it < 32; ++it) {
    v8bf lo = *(const v8bf*)&As[arow * 1024 + it * 32 + akh];
    v8bf hi = *(const v8bf*)&As[arow * 1024 + it * 32 + 8 + akh];
    v16bf af;
#pragma unroll
    for (int i = 0; i < 8; ++i) { af[i] = lo[i]; af[8 + i] = hi[i]; }

    const float* bp = brow + it * 32;
    __builtin_prefetch(bp + 64, 0, 0);         // global_prefetch_b8
    const float4 f0 = ((const float4*)bp)[0];
    const float4 f1 = ((const float4*)bp)[1];
    const float4 f2 = ((const float4*)bp)[2];
    const float4 f3 = ((const float4*)bp)[3];
    const float fv[16] = {f0.x, f0.y, f0.z, f0.w, f1.x, f1.y, f1.z, f1.w,
                          f2.x, f2.y, f2.z, f2.w, f3.x, f3.y, f3.z, f3.w};
    v16bf bfrag;
#pragma unroll
    for (int i = 0; i < 16; ++i) bfrag[i] = (bf16)fv[i];

    acc = __builtin_amdgcn_wmma_f32_16x16x32_bf16(
        false, af, false, bfrag, (short)0, acc, false, false);
  }

  const int mrow = m0 + (lane >> 4) * 8;
  const int n    = n0 + (lane & 15);
#pragma unroll
  for (int i = 0; i < 8; ++i)
    atomicAdd(&feat[(size_t)(e * 256 + mrow + i) * 128 + n], acc[i]);
}

// ---------------- gate FC: logits[b][e] = gpool[b] . gfw[e] + gfb ----------
__global__ __launch_bounds__(256)
void moe_gate_fc(const bf16* __restrict__ gpool, const float* __restrict__ gfw,
                 const float* __restrict__ gfb, float* __restrict__ logits) {
  __shared__ float red[256 * 4];
  const int b = blockIdx.x, t = threadIdx.x;
  float a0 = 0.f, a1 = 0.f, a2 = 0.f, a3 = 0.f;
  const bf16* gp = gpool + (size_t)b * 65536;
  for (int k = t; k < 65536; k += 256) {
    const float g = (float)gp[k];
    a0 += g * gfw[k];
    a1 += g * gfw[65536 + k];
    a2 += g * gfw[131072 + k];
    a3 += g * gfw[196608 + k];
  }
  red[t * 4 + 0] = a0; red[t * 4 + 1] = a1;
  red[t * 4 + 2] = a2; red[t * 4 + 3] = a3;
  __syncthreads();
  for (int s = 128; s > 0; s >>= 1) {
    if (t < s) {
#pragma unroll
      for (int ee = 0; ee < 4; ++ee) red[t * 4 + ee] += red[(t + s) * 4 + ee];
    }
    __syncthreads();
  }
  if (t < 4) logits[b * 4 + t] = red[t] + gfb[t];
}

// -------- softmax + top-2 + weighted combine + final 128->10 head ----------
__global__ __launch_bounds__(128)
void moe_finalize(const float* __restrict__ logits, const float* __restrict__ feat,
                  const float* __restrict__ efb, const float* __restrict__ fw,
                  const float* __restrict__ fb, float* __restrict__ out) {
  __shared__ float lg[4];
  __shared__ float fbuf[128];
  const int b = blockIdx.x, t = threadIdx.x;
  if (t < 4) lg[t] = logits[b * 4 + t];
  __syncthreads();

  int e0 = 0; float m0v = lg[0];
#pragma unroll
  for (int ee = 1; ee < 4; ++ee) if (lg[ee] > m0v) { m0v = lg[ee]; e0 = ee; }
  int e1 = (e0 == 0) ? 1 : 0; float m1v = lg[e1];
#pragma unroll
  for (int ee = 0; ee < 4; ++ee)
    if (ee != e0 && lg[ee] > m1v) { m1v = lg[ee]; e1 = ee; }

  const float p1 = expf(m1v - m0v);            // p0 == 1
  const float inv = 1.f / (1.f + p1);
  const float w0 = inv, w1 = p1 * inv;         // renormalized top-2 softmax

  const float f0 = feat[(size_t)(e0 * 256 + b) * 128 + t] + efb[e0 * 128 + t];
  const float f1 = feat[(size_t)(e1 * 256 + b) * 128 + t] + efb[e1 * 128 + t];
  fbuf[t] = w0 * f0 + w1 * f1;
  __syncthreads();

  if (t < 10) {
    float s = fb[t];
    for (int n = 0; n < 128; ++n) s += fbuf[n] * fw[t * 128 + n];
    out[b * 10 + t] = s;
  }
}

// ---------------------------------------------------------------------------
extern "C" void kernel_launch(void* const* d_in, const int* in_sizes, int n_in,
                              void* d_out, int out_size, void* d_ws, size_t ws_size,
                              hipStream_t stream) {
  (void)in_sizes; (void)n_in; (void)out_size; (void)ws_size;
  const float* x   = (const float*)d_in[0];
  const float* ew1 = (const float*)d_in[1];
  const float* eb1 = (const float*)d_in[2];
  const float* ew2 = (const float*)d_in[3];
  const float* eb2 = (const float*)d_in[4];
  const float* efw = (const float*)d_in[5];
  const float* efb = (const float*)d_in[6];
  const float* gw  = (const float*)d_in[7];
  const float* gb  = (const float*)d_in[8];
  const float* gfw = (const float*)d_in[9];
  const float* gfb = (const float*)d_in[10];
  const float* fw  = (const float*)d_in[11];
  const float* fb  = (const float*)d_in[12];
  float* out = (float*)d_out;

  char* ws = (char*)d_ws;
  bf16*  a1     = (bf16*)(ws + OFF_A1);
  bf16*  h2     = (bf16*)(ws + OFF_H2);
  bf16*  bmat2  = (bf16*)(ws + OFF_BMAT2);
  bf16*  gpool  = (bf16*)(ws + OFF_GPOOL);
  float* feat   = (float*)(ws + OFF_FEAT);
  float* logits = (float*)(ws + OFF_LOG);

  hipMemsetAsync(feat, 0, (size_t)4 * 256 * 128 * sizeof(float), stream);
  moe_prep_b2<<<288, 256, 0, stream>>>(ew2, bmat2);
  moe_expert_conv1<<<131072, 256, 0, stream>>>(x, ew1, eb1, a1);
  moe_gate_conv<<<65536, 256, 0, stream>>>(x, gw, gb, gpool);
  moe_conv2_wmma<<<dim3(2048, 4), 256, 0, stream>>>(a1, bmat2, eb2, h2);
  moe_fc_wmma<<<dim3(16, 16, 4), 256, 0, stream>>>(h2, efw, feat);
  moe_gate_fc<<<256, 256, 0, stream>>>(gpool, gfw, gfb, logits);
  moe_finalize<<<256, 128, 0, stream>>>(logits, feat, efb, fw, fb, out);
}
